// YatPerformerCausalAttention_56367150793413
// MI455X (gfx1250) — compile-verified, matrix operand order checked by reading, not run on hardware
//
#include <hip/hip_runtime.h>
#include <hip/hip_bf16.h>

// ---------------------------------------------------------------------------
// Types for CDNA5 WMMA (wave32): v_wmma_f32_16x16x32_bf16
// ---------------------------------------------------------------------------
typedef __bf16 v16bf __attribute__((ext_vector_type(16)));
typedef __bf16 v8bf  __attribute__((ext_vector_type(8)));
typedef float  v8f   __attribute__((ext_vector_type(8)));

#define DEV __device__ __forceinline__

// Problem constants
constexpr int Bn   = 2;
constexpr int T    = 4096;
constexpr int C    = 1024;
constexpr int H    = 16;
constexpr int D    = 64;
constexpr int M    = 32;
constexpr int R    = 2;
constexpr int F    = 64;     // R*M
constexpr int BH   = Bn * H; // 32
constexpr int LC   = 64;     // internal chunk length
constexpr int NC   = T / LC; // 64 chunks

// ---------------------------------------------------------------------------
// Load a 16x32 bf16 A/B fragment (K-contiguous memory) into the documented
// CDNA5 VGPR layout: lanes 0-15 = rows, lane halves split K:
//   elems 0..7  -> K = 8*(lane>>4) + 0..7
//   elems 8..15 -> K = 16 + 8*(lane>>4) + 0..7
// ---------------------------------------------------------------------------
DEV v16bf load_frag(const __bf16* __restrict__ p, int row, int stride, int k0,
                    int lane) {
  const int half = (lane >> 4) & 1;
  const __bf16* base = p + (row + (lane & 15)) * stride + (k0 + half * 8);
  v16bf f;
  *reinterpret_cast<v8bf*>(&f)       = *reinterpret_cast<const v8bf*>(base);
  *(reinterpret_cast<v8bf*>(&f) + 1) = *reinterpret_cast<const v8bf*>(base + 16);
  return f;
}

DEV v8f wmma_bf16(v16bf a, v16bf b, v8f c) {
  // (neg_a, A, neg_b, B, c_mod, C, reuse_a, reuse_b)
  return __builtin_amdgcn_wmma_f32_16x16x32_bf16(false, a, false, b, (short)0,
                                                 c, false, false);
}

// ---------------------------------------------------------------------------
// K0a: f32 -> bf16 conversion
// ---------------------------------------------------------------------------
__global__ void cvt_bf16_kernel(const float* __restrict__ src,
                                __bf16* __restrict__ dst, int n) {
  int i = blockIdx.x * blockDim.x + threadIdx.x;
  if (i < n) dst[i] = (__bf16)src[i];
}

// ---------------------------------------------------------------------------
// K0b: omega [R,H,D,M] f32 -> omT [H,F,D] bf16  (f = r*M+m, d contiguous)
// ---------------------------------------------------------------------------
__global__ void omega_cvt_kernel(const float* __restrict__ omega,
                                 __bf16* __restrict__ omT) {
  const int i = blockIdx.x * 256 + threadIdx.x;  // over H*F*D = 65536
  if (i >= H * F * D) return;
  const int d = i & 63;
  const int f = (i >> 6) & 63;
  const int h = i >> 12;
  const int r = f >> 5, m = f & 31;
  omT[i] = (__bf16)omega[(((r * H + h) * D) + d) * M + m];
}

// ---------------------------------------------------------------------------
// K1: QKV GEMM.  qkv[8192,3072] = x_bf[8192,1024] @ wqkv_bf[3072,1024]^T + b
// Block: 256 thr = 8 waves (2 M-strips x 4 N-strips); block tile 128 x 256;
// wave tile 64x64 (16 WMMAs per 8 fragment loads -> 0.5 loads/WMMA).
// Epilogue: one base pointer per 16x16 tile, immediate-offset stores
// (keeps the 128 accumulator VGPRs from spilling to scratch).
// ---------------------------------------------------------------------------
__global__ __launch_bounds__(256) void gemm_qkv_kernel(
    const __bf16* __restrict__ A,    // [8192,1024]
    const __bf16* __restrict__ Bw,   // [3072,1024]
    const float* __restrict__ bias,  // [3072]
    float* __restrict__ qbuf, float* __restrict__ kbuf,
    __bf16* __restrict__ vdt) {
  const int lane = threadIdx.x & 31;
  const int wave = threadIdx.x >> 5;
  const int wm = wave & 1, wn = wave >> 1;
  const int m0 = blockIdx.x * 128 + wm * 64;
  const int n0 = blockIdx.y * 256 + wn * 64;

  v8f acc[4][4] = {};
  for (int k0 = 0; k0 < C; k0 += 32) {
    v16bf a[4], b[4];
#pragma unroll
    for (int i = 0; i < 4; ++i) a[i] = load_frag(A, m0 + 16 * i, C, k0, lane);
#pragma unroll
    for (int j = 0; j < 4; ++j) b[j] = load_frag(Bw, n0 + 16 * j, C, k0, lane);
#pragma unroll
    for (int i = 0; i < 4; ++i)
#pragma unroll
      for (int j = 0; j < 4; ++j) acc[i][j] = wmma_bf16(a[i], b[j], acc[i][j]);
  }

  const int half = lane >> 4, nn = lane & 15;
  const int b_ = m0 >> 12;   // batch index: uniform (tiles are 4096-aligned)
  const int tb = m0 & 4095;  // token base
#pragma unroll
  for (int j = 0; j < 4; ++j) {
    const int ncol = n0 + j * 16;       // 16-aligned -> part/h/dbase uniform
    const float bv = bias[ncol + nn];
    const int part = ncol >> 10;        // 0=q 1=k 2=v
    const int cc = ncol & 1023;
    const int h = cc >> 6, dbase = cc & 63;
    const int bh = b_ * H + h;
#pragma unroll
    for (int i = 0; i < 4; ++i) {
      const int t0r = tb + i * 16 + 8 * half;  // + g
      if (part == 2) {
        __bf16* p = vdt + ((bh << 6) + dbase + nn) * T + t0r;
#pragma unroll
        for (int g = 0; g < 8; ++g) p[g] = (__bf16)(acc[i][j][g] + bv);
      } else {
        float* p = (part == 0 ? qbuf : kbuf) + ((bh * T + t0r) << 6) + dbase + nn;
#pragma unroll
        for (int g = 0; g < 8; ++g) p[g << 6] = acc[i][j][g] + bv;
      }
    }
  }
}

// ---------------------------------------------------------------------------
// K2: feature map (WMMA). One block = 64 tokens of one (b,h).
//   zn = z / max(||z||,1e-12)  (f32)  -> bf16 LDS tile [64 x 64]
//   proj = zn @ omT[h]^T via WMMA (M=64 tok, N=64 feat, K=64)
//   elementwise poly*prf*sqw tail in VALU, write bf16 [BH,T,F] (+[BH,F,T]).
// ---------------------------------------------------------------------------
__global__ __launch_bounds__(256) void features_kernel(
    const float* __restrict__ z,     // [BH,T,64]
    const __bf16* __restrict__ omT,  // [H,64,64]
    const float* __restrict__ qn, const float* __restrict__ qw,
    __bf16* __restrict__ feat, __bf16* __restrict__ feat_t) {
  __shared__ float  zf[64 * 64];
  __shared__ __bf16 znb[64 * 64];
  const int tile = blockIdx.x;
  const int bh = tile >> 6;
  const int t0 = (tile & 63) << 6;
  const int h = bh & (H - 1);
  const int tid = threadIdx.x, lane = tid & 31, wave = tid >> 5;

  // stage z tile (coalesced)
  for (int i = 0; i < 16; ++i) {
    const int idx = i * 256 + tid;
    zf[idx] = z[(bh * T + t0) * 64 + idx];
  }
  __syncthreads();
  // normalize rows in f32, emit bf16
  if (tid < 64) {
    float ss = 0.f;
    for (int j = 0; j < 64; ++j) { float v = zf[tid * 64 + j]; ss += v * v; }
    const float inv = 1.0f / fmaxf(sqrtf(ss), 1e-12f);
    for (int j = 0; j < 64; ++j) znb[tid * 64 + j] = (__bf16)(zf[tid * 64 + j] * inv);
  }
  __syncthreads();

  // proj GEMM: 16 tiles over (token,f); 8 waves -> 2 tiles each
  const int mi = wave & 3, nb = (wave >> 2) * 2;
  const __bf16* ob = omT + h * (F * D);
  v8f acc[2] = {};
  for (int d0 = 0; d0 < D; d0 += 32) {
    v16bf a  = load_frag(&znb[0], mi * 16, 64, d0, lane);
    v16bf b0 = load_frag(ob, nb * 16, D, d0, lane);
    v16bf b1 = load_frag(ob, nb * 16 + 16, D, d0, lane);
    acc[0] = wmma_bf16(a, b0, acc[0]);
    acc[1] = wmma_bf16(a, b1, acc[1]);
  }

  const float inv_sqrt_m = 0.17677669529663687f;  // 1/sqrt(32)
  const int half = lane >> 4, nn = lane & 15;
  for (int j = 0; j < 2; ++j) {
    const int fcol = (nb + j) * 16 + nn;
    const int r = fcol >> 5;
    const float s = qn[r];
    const float sqrt2s = sqrtf(2.0f * fmaxf(s, 0.0f));
    const float sqw = sqrtf(fmaxf(qw[r], 0.0f));
#pragma unroll
    for (int g = 0; g < 8; ++g) {
      const int trow = mi * 16 + g + 8 * half;
      const float p = acc[j][g];
      const float poly = p * p * inv_sqrt_m;
      const float arg = fminf(fmaxf(p * sqrt2s - s, -20.0f), 20.0f);
      const float val = poly * (__expf(arg) * inv_sqrt_m) * sqw;
      const int tg = t0 + trow;
      feat[(bh * T + tg) * F + fcol] = (__bf16)val;
      if (feat_t) feat_t[(bh * F + fcol) * T + tg] = (__bf16)val;
    }
  }
}

// ---------------------------------------------------------------------------
// K3: per-chunk state  KV_c^T[d,f] = sum_{t in chunk} v[t,d]*kf[t,f]
// GEMM: A = v^T[D,T] (t-contig), B = kf^T[F,T] (t-contig), K = 64 tokens.
// ---------------------------------------------------------------------------
__global__ __launch_bounds__(256) void attn_state_kernel(
    const __bf16* __restrict__ vdt,  // [BH,D,T]
    const __bf16* __restrict__ kft,  // [BH,F,T]
    float* __restrict__ kvc,         // [BH,NC,D,F]
    float* __restrict__ ksc) {       // [BH,NC,F]
  const int c = blockIdx.x, bh = blockIdx.y;
  const int tid = threadIdx.x, lane = tid & 31, wave = tid >> 5;
  const __bf16* vb = vdt + (size_t)bh * D * T;
  const __bf16* kb = kft + (size_t)bh * F * T;
  const int mi = wave & 3, nb = (wave >> 2) * 2;

  v8f acc[2] = {};
  for (int kk = 0; kk < LC; kk += 32) {
    const int k0 = c * LC + kk;
    v16bf a  = load_frag(vb, mi * 16, T, k0, lane);
    v16bf b0 = load_frag(kb, nb * 16, T, k0, lane);
    v16bf b1 = load_frag(kb, nb * 16 + 16, T, k0, lane);
    acc[0] = wmma_bf16(a, b0, acc[0]);
    acc[1] = wmma_bf16(a, b1, acc[1]);
  }
  float* outp = kvc + ((size_t)(bh * NC + c) << 12);
  const int half = lane >> 4, nn = lane & 15;
  for (int j = 0; j < 2; ++j) {
    float* p = outp + ((mi * 16 + 8 * half) << 6) + (nb + j) * 16 + nn;
#pragma unroll
    for (int g = 0; g < 8; ++g) p[g << 6] = acc[j][g];
  }
  if (tid < 64) {
    float s = 0.f;
    const __bf16* kr = kb + tid * T + c * LC;
    for (int i = 0; i < LC; ++i) s += (float)kr[i];
    ksc[((bh * NC + c) << 6) + tid] = s;
  }
}

// ---------------------------------------------------------------------------
// K4: exclusive prefix over chunks (f32), emit bf16 carry-in state.
// ---------------------------------------------------------------------------
__global__ __launch_bounds__(256) void attn_prefix_kernel(
    const float* __restrict__ kvc, __bf16* __restrict__ kvp,
    const float* __restrict__ ksc, float* __restrict__ ksp) {
  const int bh = blockIdx.x, tid = threadIdx.x;
  float carry[16];
#pragma unroll
  for (int e = 0; e < 16; ++e) carry[e] = 0.f;
  const int off = tid * 16;
  for (int c = 0; c < NC; ++c) {
    const size_t base = ((size_t)(bh * NC + c) << 12) + off;
#pragma unroll
    for (int e = 0; e < 16; ++e) kvp[base + e] = (__bf16)carry[e];
#pragma unroll
    for (int e = 0; e < 16; ++e) carry[e] += kvc[base + e];
  }
  if (tid < 64) {
    float kc = 0.f;
    for (int c = 0; c < NC; ++c) {
      const size_t b2 = ((size_t)(bh * NC + c) << 6) + tid;
      ksp[b2] = kc;
      kc += ksc[b2];
    }
  }
}

// ---------------------------------------------------------------------------
// K5: chunk output.
//   S = qf kf^T (masked, intra-chunk), den = rowsum(S) + qf.ksum_in
//   O = qf @ KV_in + S @ v ; out = O/den -> attn bf16 [B,T,C]
// ---------------------------------------------------------------------------
__global__ __launch_bounds__(256) void attn_out_kernel(
    const __bf16* __restrict__ qf,   // [BH,T,F]
    const __bf16* __restrict__ kf,   // [BH,T,F]
    const __bf16* __restrict__ vdt,  // [BH,D,T]
    const __bf16* __restrict__ kvp,  // [BH,NC,D,F]
    const float* __restrict__ ksp,   // [BH,NC,F]
    __bf16* __restrict__ attn) {     // [B,T,C]
  __shared__ float  Sf[64 * 64];
  __shared__ __bf16 Sb[64 * 64];
  __shared__ float  den[64];
  const int c = blockIdx.x, bh = blockIdx.y;
  const int b = bh >> 4, h = bh & 15;
  const int tid = threadIdx.x, lane = tid & 31, wave = tid >> 5;
  const int mi = wave & 3, nb = (wave >> 2) * 2;
  const int half = lane >> 4, nn = lane & 15;
  const __bf16* qb  = qf + (size_t)bh * T * F;
  const __bf16* kb  = kf + (size_t)bh * T * F;
  const __bf16* vb  = vdt + (size_t)bh * D * T;
  const __bf16* kvb = kvp + ((size_t)(bh * NC + c) << 12);
  const float*  ksb = ksp + ((size_t)(bh * NC + c) << 6);
  const int tc0 = c * LC;

  // 1) masked score tile -> LDS (f32 for den, bf16 as next A operand)
  {
    v8f s[2] = {};
    for (int f0 = 0; f0 < F; f0 += 32) {
      v16bf a  = load_frag(qb, tc0 + mi * 16, F, f0, lane);
      v16bf b0 = load_frag(kb, tc0 + nb * 16, F, f0, lane);
      v16bf b1 = load_frag(kb, tc0 + nb * 16 + 16, F, f0, lane);
      s[0] = wmma_bf16(a, b0, s[0]);
      s[1] = wmma_bf16(a, b1, s[1]);
    }
    for (int j = 0; j < 2; ++j) {
      const int ni = nb + j;
#pragma unroll
      for (int g = 0; g < 8; ++g) {
        const int tr = mi * 16 + g + 8 * half;
        const int tc = ni * 16 + nn;
        float v = s[j][g];
        if (tc > tr) v = 0.0f;  // causal mask (inclusive diagonal kept)
        Sf[tr * 64 + tc] = v;
        Sb[tr * 64 + tc] = (__bf16)v;
      }
    }
  }
  __syncthreads();

  // 2) denominator
  if (tid < 64) {
    float ssum = 0.f;
    for (int j = 0; j < 64; ++j) ssum += Sf[tid * 64 + j];
    const __bf16* qrow = qb + (size_t)(tc0 + tid) * F;
    float dd = 0.f;
    for (int f = 0; f < F; ++f) dd += (float)qrow[f] * ksb[f];
    den[tid] = fmaxf(ssum + dd, 1e-6f);
  }
  __syncthreads();

  // 3) output: carry GEMM + intra-chunk GEMM
  {
    v8f o[2] = {};
    for (int f0 = 0; f0 < F; f0 += 32) {  // carry: O += qf @ KV_in
      v16bf a  = load_frag(qb, tc0 + mi * 16, F, f0, lane);
      v16bf b0 = load_frag(kvb, nb * 16, F, f0, lane);
      v16bf b1 = load_frag(kvb, nb * 16 + 16, F, f0, lane);
      o[0] = wmma_bf16(a, b0, o[0]);
      o[1] = wmma_bf16(a, b1, o[1]);
    }
    const __bf16* sl = &Sb[0];
    for (int kk = 0; kk < LC; kk += 32) {  // intra: O += S @ v
      v16bf a  = load_frag(sl, mi * 16, 64, kk, lane);
      v16bf b0 = load_frag(vb, nb * 16, T, tc0 + kk, lane);
      v16bf b1 = load_frag(vb, nb * 16 + 16, T, tc0 + kk, lane);
      o[0] = wmma_bf16(a, b0, o[0]);
      o[1] = wmma_bf16(a, b1, o[1]);
    }
    for (int j = 0; j < 2; ++j) {
      const int dbase = (nb + j) * 16;
#pragma unroll
      for (int g = 0; g < 8; ++g) {
        const int trl = mi * 16 + g + 8 * half;
        const int tg = tc0 + trl;
        const float v = o[j][g] / den[trl];
        attn[((size_t)(b * T + tg) << 10) + (h << 6) + dbase + nn] = (__bf16)v;
      }
    }
  }
}

// ---------------------------------------------------------------------------
// K6: output projection. out[8192,1024] = attn_bf @ wout_bf^T + b_out (f32)
// Same 128x256 block / 64x64 wave tiling as K1; non-temporal final stores.
// ---------------------------------------------------------------------------
__global__ __launch_bounds__(256) void gemm_out_kernel(
    const __bf16* __restrict__ A,    // [8192,1024]
    const __bf16* __restrict__ Bw,   // [1024,1024]
    const float* __restrict__ bias,  // [1024]
    float* __restrict__ out) {       // [8192,1024]
  const int lane = threadIdx.x & 31;
  const int wave = threadIdx.x >> 5;
  const int wm = wave & 1, wn = wave >> 1;
  const int m0 = blockIdx.x * 128 + wm * 64;
  const int n0 = blockIdx.y * 256 + wn * 64;

  v8f acc[4][4] = {};
  for (int k0 = 0; k0 < C; k0 += 32) {
    v16bf a[4], b[4];
#pragma unroll
    for (int i = 0; i < 4; ++i) a[i] = load_frag(A, m0 + 16 * i, C, k0, lane);
#pragma unroll
    for (int j = 0; j < 4; ++j) b[j] = load_frag(Bw, n0 + 16 * j, C, k0, lane);
#pragma unroll
    for (int i = 0; i < 4; ++i)
#pragma unroll
      for (int j = 0; j < 4; ++j) acc[i][j] = wmma_bf16(a[i], b[j], acc[i][j]);
  }
  const int half = lane >> 4, nn = lane & 15;
#pragma unroll
  for (int j = 0; j < 4; ++j) {
    const float bv = bias[n0 + j * 16 + nn];
#pragma unroll
    for (int i = 0; i < 4; ++i) {
      float* p = out + (size_t)(m0 + i * 16 + 8 * half) * C + n0 + j * 16 + nn;
#pragma unroll
      for (int g = 0; g < 8; ++g)
        __builtin_nontemporal_store(acc[i][j][g] + bv, p + (size_t)g * C);
    }
  }
}

// ---------------------------------------------------------------------------
// Host-side launcher
// ---------------------------------------------------------------------------
extern "C" void kernel_launch(void* const* d_in, const int* in_sizes, int n_in,
                              void* d_out, int out_size, void* d_ws,
                              size_t ws_size, hipStream_t stream) {
  const float* x     = (const float*)d_in[0];  // [B,T,C]
  const float* w_qkv = (const float*)d_in[1];  // [3C,C]
  const float* b_qkv = (const float*)d_in[2];  // [3C]
  const float* w_out = (const float*)d_in[3];  // [C,C]
  const float* b_out = (const float*)d_in[4];  // [C]
  const float* omega = (const float*)d_in[5];  // [R,H,D,M]
  const float* qn    = (const float*)d_in[6];  // [R]
  const float* qw    = (const float*)d_in[7];  // [R]
  float* out = (float*)d_out;

  // Workspace carve-up (256B aligned)
  char* ws = (char*)d_ws;
  size_t off = 0;
  auto alloc = [&](size_t bytes) {
    char* p = ws + off;
    off = (off + bytes + 255) & ~(size_t)255;
    return p;
  };
  const size_t MT = (size_t)Bn * T;  // 8192 rows
  __bf16* x_bf    = (__bf16*)alloc(MT * C * 2);
  __bf16* wqkv_bf = (__bf16*)alloc((size_t)3 * C * C * 2);
  __bf16* wout_bf = (__bf16*)alloc((size_t)C * C * 2);
  __bf16* omT     = (__bf16*)alloc((size_t)H * F * D * 2);
  float*  qbuf    = (float*)alloc((size_t)BH * T * D * 4);
  float*  kbuf    = (float*)alloc((size_t)BH * T * D * 4);
  __bf16* vdt     = (__bf16*)alloc((size_t)BH * D * T * 2);
  __bf16* qf      = (__bf16*)alloc((size_t)BH * T * F * 2);
  __bf16* kf      = (__bf16*)alloc((size_t)BH * T * F * 2);
  __bf16* kft     = (__bf16*)alloc((size_t)BH * F * T * 2);
  float*  kvc     = (float*)alloc((size_t)BH * NC * D * F * 4);
  __bf16* kvp     = (__bf16*)alloc((size_t)BH * NC * D * F * 2);
  float*  ksc     = (float*)alloc((size_t)BH * NC * F * 4);
  float*  ksp     = (float*)alloc((size_t)BH * NC * F * 4);
  __bf16* attn    = (__bf16*)alloc(MT * C * 2);
  (void)ws_size; (void)n_in; (void)in_sizes; (void)out_size;

  // K0: conversions
  {
    int n1 = (int)(MT * C);
    cvt_bf16_kernel<<<(n1 + 255) / 256, 256, 0, stream>>>(x, x_bf, n1);
    int n2 = 3 * C * C;
    cvt_bf16_kernel<<<(n2 + 255) / 256, 256, 0, stream>>>(w_qkv, wqkv_bf, n2);
    int n3 = C * C;
    cvt_bf16_kernel<<<(n3 + 255) / 256, 256, 0, stream>>>(w_out, wout_bf, n3);
    omega_cvt_kernel<<<(H * F * D + 255) / 256, 256, 0, stream>>>(omega, omT);
  }
  // K1: QKV projection (WMMA, 128x256 block tile)
  gemm_qkv_kernel<<<dim3(MT / 128, (3 * C) / 256), 256, 0, stream>>>(
      x_bf, wqkv_bf, b_qkv, qbuf, kbuf, vdt);
  // K2: features for q and k (WMMA projection)
  features_kernel<<<BH * (T / 64), 256, 0, stream>>>(qbuf, omT, qn, qw, qf,
                                                     (__bf16*)nullptr);
  features_kernel<<<BH * (T / 64), 256, 0, stream>>>(kbuf, omT, qn, qw, kf,
                                                     kft);
  // K3: per-chunk KV^T states (WMMA)
  attn_state_kernel<<<dim3(NC, BH), 256, 0, stream>>>(vdt, kft, kvc, ksc);
  // K4: exclusive prefix over chunks
  attn_prefix_kernel<<<BH, 256, 0, stream>>>(kvc, kvp, ksc, ksp);
  // K5: chunk outputs (WMMA: scores + carry + intra)
  attn_out_kernel<<<dim3(NC, BH), 256, 0, stream>>>(qf, kf, vdt, kvp, ksp,
                                                    attn);
  // K6: output projection (WMMA)
  gemm_out_kernel<<<dim3(MT / 128, C / 256), 256, 0, stream>>>(attn, wout_bf,
                                                               b_out, out);
}
